// SelfAttention_6682969112792
// MI455X (gfx1250) — matrix operand order repaired and checked by reading, compiled, not run on hardware
//
#include <hip/hip_runtime.h>
#include <hip/hip_bf16.h>

typedef __attribute__((ext_vector_type(16))) __bf16 v16bf;
typedef __attribute__((ext_vector_type(8)))  float  v8f;

#define B_  4
#define S_  2048
#define H_  2048
#define NH_ 16
#define HD_ 128

union FragBF {
  v16bf  v;
  __bf16 h[16];
  uint4  q[2];
};
static_assert(sizeof(FragBF) == 32, "frag size");

// ---------------------------------------------------------------------------
// fp32 -> bf16 pre-conversion (vectorized: float4 in, 4xbf16 packed out)
// ---------------------------------------------------------------------------
__global__ __launch_bounds__(256) void f32_to_bf16_kernel(
    const float* __restrict__ src, __bf16* __restrict__ dst, long n4) {
  const long i = (long)blockIdx.x * 256 + threadIdx.x;  // float4 index
  if (i >= n4) return;
  const float4 f = ((const float4*)src)[i];
  union { __bf16 h[4]; uint2 u; } pk;
  pk.h[0] = (__bf16)f.x; pk.h[1] = (__bf16)f.y;
  pk.h[2] = (__bf16)f.z; pk.h[3] = (__bf16)f.w;
  ((uint2*)dst)[i] = pk.u;
}

// ---------------------------------------------------------------------------
// C[M,N] = A[M,K] * W[N,K]^T   (all-bf16 operands, fp32 accumulate)
// Wave tile 32x64 (2x4 WMMA frags), block 8 waves -> 64x256 tile.
// MODE 0: store fp32 row-major [M,N]
// MODE 1: store bf16 scattered into qkv[3][B][NH][S][HD]
// ---------------------------------------------------------------------------
template <int MODE>
__global__ __launch_bounds__(256) void gemm_wmma_kernel(
    const __bf16* __restrict__ A, const __bf16* __restrict__ W,
    void* __restrict__ Cp, int M, int N, int K) {
  const int lane = threadIdx.x & 31;
  const int wave = threadIdx.x >> 5;
  const int wm = wave & 1, wn = wave >> 1;
  const int row0 = blockIdx.x * 64  + wm * 32;
  const int col0 = blockIdx.y * 256 + wn * 64;
  const int lh  = lane >> 4;   // 16-lane half
  const int l16 = lane & 15;

  v8f acc[2][4] = {};

  for (int k0 = 0; k0 < K; k0 += 32) {
    FragBF af[2], bf[4];
    // A fragment (16x32): lane = M row; elems 0..7 at k0+8*lh, elems 8..15 at +16
#pragma unroll
    for (int i = 0; i < 2; ++i) {
      const __bf16* p = A + (size_t)(row0 + i * 16 + l16) * K + k0 + 8 * lh;
      af[i].q[0] = *(const uint4*)(p);
      af[i].q[1] = *(const uint4*)(p + 16);
    }
    // B fragment (32x16): lane = N col; elems 0..7 at k0+8*lh, elems 8..15 at +16
#pragma unroll
    for (int j = 0; j < 4; ++j) {
      const __bf16* p = W + (size_t)(col0 + j * 16 + l16) * K + k0 + 8 * lh;
      bf[j].q[0] = *(const uint4*)(p);
      bf[j].q[1] = *(const uint4*)(p + 16);
    }
#pragma unroll
    for (int i = 0; i < 2; ++i)
#pragma unroll
      for (int j = 0; j < 4; ++j)
        acc[i][j] = __builtin_amdgcn_wmma_f32_16x16x32_bf16(
            false, af[i].v, false, bf[j].v, (short)0, acc[i][j], false, false);
  }

  // epilogue: C/D layout m = r + 8*lh, n = l16
  if constexpr (MODE == 0) {
    float* C = (float*)Cp;
#pragma unroll
    for (int i = 0; i < 2; ++i)
#pragma unroll
      for (int j = 0; j < 4; ++j)
#pragma unroll
        for (int r = 0; r < 8; ++r) {
          const int m = r + 8 * lh;
          C[(size_t)(row0 + i * 16 + m) * N + (col0 + j * 16 + l16)] = acc[i][j][r];
        }
  } else {
    __bf16* C = (__bf16*)Cp;
#pragma unroll
    for (int j = 0; j < 4; ++j) {
      const int colb  = col0 + j * 16;
      const int three = colb / H_;
      const int rem   = colb % H_;
      const int nh    = rem / HD_;
      const int hd    = rem % HD_ + l16;
#pragma unroll
      for (int i = 0; i < 2; ++i)
#pragma unroll
        for (int r = 0; r < 8; ++r) {
          const int m   = r + 8 * lh;
          const int row = row0 + i * 16 + m;
          const int bb  = row / S_;
          const int s   = row % S_;
          const size_t off =
              ((((size_t)three * B_ + bb) * NH_ + nh) * S_ + s) * HD_ + hd;
          C[off] = (__bf16)acc[i][j][r];
        }
    }
  }
}

// ---------------------------------------------------------------------------
// RoPE in-place on q (three=0) and k (three=1) parts of qkv
// ---------------------------------------------------------------------------
__global__ __launch_bounds__(256) void rope_kernel(__bf16* qkv,
                                                   const float* __restrict__ freqs) {
  const size_t total = (size_t)2 * B_ * NH_ * S_ * (HD_ / 2);
  size_t idx = (size_t)blockIdx.x * 256 + threadIdx.x;
  if (idx >= total) return;
  const int i2 = (int)(idx & 63);          // HD/2 = 64
  size_t t = idx >> 6;
  const int s = (int)(t & (S_ - 1)); t >>= 11;
  const int nh = (int)(t & 15);      t >>= 4;
  const int bb = (int)(t & 3);       t >>= 2;
  const int three = (int)t;          // 0 or 1
  const size_t base =
      ((((size_t)three * B_ + bb) * NH_ + nh) * S_ + s) * HD_ + 2 * i2;
  const float f = freqs[s * (HD_ / 2) + i2];
  const float c = cosf(f), sn = sinf(f);
  const float tr = (float)qkv[base], ti = (float)qkv[base + 1];
  qkv[base]     = (__bf16)(tr * c - ti * sn);
  qkv[base + 1] = (__bf16)(tr * sn + ti * c);
}

// ---------------------------------------------------------------------------
// Flash attention: one wave per (b, head, 16-query tile). Causal.
// ---------------------------------------------------------------------------
__global__ __launch_bounds__(32) void attn_kernel(const __bf16* __restrict__ qkv,
                                                  __bf16* __restrict__ aout) {
  __shared__ __align__(16) __bf16 Plds[16 * 32];   // P tile, A-fragment staging
  __shared__ __align__(16) __bf16 Vt[128 * 32];    // V tile transposed [col][key]

  const int lane = threadIdx.x;
  const int l16 = lane & 15, lh = lane >> 4;
  const int qt = blockIdx.x, nh = blockIdx.y, b = blockIdx.z;
  const int q0 = qt * 16;

  const size_t headoff = ((size_t)b * NH_ + nh) * S_ * HD_;
  const size_t planesz = (size_t)B_ * NH_ * S_ * HD_;
  const __bf16* Q  = qkv + headoff;             // three = 0
  const __bf16* Kp = qkv + planesz + headoff;   // three = 1
  const __bf16* Vp = qkv + 2 * planesz + headoff;

  // Q A-fragments (held for whole loop): 16 rows x 128 head-dim
  FragBF qf[4];
#pragma unroll
  for (int kk = 0; kk < 4; ++kk) {
    const __bf16* p = Q + (size_t)(q0 + l16) * HD_ + kk * 32 + 8 * lh;
    qf[kk].q[0] = *(const uint4*)(p);
    qf[kk].q[1] = *(const uint4*)(p + 16);
  }

  float rmax[8], rsum[8];
  v8f o[8] = {};
#pragma unroll
  for (int r = 0; r < 8; ++r) { rmax[r] = -INFINITY; rsum[r] = 0.f; }

  const float scale = 0.08838834764831845f;  // 1/sqrt(128)
  const int nkt = (q0 + 15) / 32 + 1;

  for (int kt = 0; kt < nkt; ++kt) {
    const int k0 = kt * 32;

    // ---- scores: two 16x16 tiles (key halves), K-dim 128 = 4 chained WMMAs
    v8f acc0 = {}, acc1 = {};
#pragma unroll
    for (int hv = 0; hv < 2; ++hv) {
#pragma unroll
      for (int kk = 0; kk < 4; ++kk) {
        FragBF kf;
        const __bf16* p = Kp + (size_t)(k0 + hv * 16 + l16) * HD_ + kk * 32 + 8 * lh;
        kf.q[0] = *(const uint4*)(p);
        kf.q[1] = *(const uint4*)(p + 16);
        if (hv == 0)
          acc0 = __builtin_amdgcn_wmma_f32_16x16x32_bf16(false, qf[kk].v, false,
                                                         kf.v, (short)0, acc0, false, false);
        else
          acc1 = __builtin_amdgcn_wmma_f32_16x16x32_bf16(false, qf[kk].v, false,
                                                         kf.v, (short)0, acc1, false, false);
      }
    }

    // ---- causal mask + online softmax (row stats via 16-lane butterflies)
    float corr[8];
#pragma unroll
    for (int r = 0; r < 8; ++r) {
      const int m = r + 8 * lh;
      const int qrow = q0 + m;
      float s0 = acc0[r] * scale;
      float s1 = acc1[r] * scale;
      if (k0 + l16 > qrow)      s0 = -INFINITY;
      if (k0 + 16 + l16 > qrow) s1 = -INFINITY;
      float tm = fmaxf(s0, s1);
#pragma unroll
      for (int off = 1; off < 16; off <<= 1) tm = fmaxf(tm, __shfl_xor(tm, off, 32));
      const float nm = fmaxf(rmax[r], tm);
      const float p0 = __expf(s0 - nm);
      const float p1 = __expf(s1 - nm);
      corr[r] = __expf(rmax[r] - nm);
      rmax[r] = nm;
      float ts = p0 + p1;
#pragma unroll
      for (int off = 1; off < 16; off <<= 1) ts += __shfl_xor(ts, off, 32);
      rsum[r] = rsum[r] * corr[r] + ts;
      Plds[m * 32 + l16]      = (__bf16)p0;
      Plds[m * 32 + 16 + l16] = (__bf16)p1;
    }
#pragma unroll
    for (int nb = 0; nb < 8; ++nb)
#pragma unroll
      for (int r = 0; r < 8; ++r) o[nb][r] *= corr[r];

    // ---- transpose V tile (32 keys x 128 cols) into Vt[col][key] (bf16)
    {
      const int kA = k0 + 2 * l16;                 // this lane owns keys kA, kA+1
      const __bf16* vA = Vp + (size_t)kA * HD_ + lh * 64;  // half the columns
      const __bf16* vB = vA + HD_;
      unsigned* Vtd = (unsigned*)Vt;
#pragma unroll
      for (int ch = 0; ch < 4; ++ch) {
        uint4 ra0 = *(const uint4*)(vA + ch * 16);
        uint4 ra1 = *(const uint4*)(vA + ch * 16 + 8);
        uint4 rb0 = *(const uint4*)(vB + ch * 16);
        uint4 rb1 = *(const uint4*)(vB + ch * 16 + 8);
        unsigned a4[8] = {ra0.x, ra0.y, ra0.z, ra0.w, ra1.x, ra1.y, ra1.z, ra1.w};
        unsigned b4[8] = {rb0.x, rb0.y, rb0.z, rb0.w, rb1.x, rb1.y, rb1.z, rb1.w};
#pragma unroll
        for (int tix = 0; tix < 8; ++tix) {
          const int c0 = lh * 64 + ch * 16 + 2 * tix;
          const unsigned av = a4[tix], bv = b4[tix];
          const unsigned lo = (av & 0xffffu) | (bv << 16);      // col c0  : keys kA,kA+1
          const unsigned hi = (av >> 16) | (bv & 0xffff0000u);  // col c0+1: keys kA,kA+1
          Vtd[c0 * 16 + l16]       = lo;
          Vtd[(c0 + 1) * 16 + l16] = hi;
        }
      }
    }
    __syncthreads();  // single-wave workgroup: S_NOP in HW, compiler fence

    // ---- P A-fragment from LDS, then O += P x V (8 col blocks)
    FragBF pf;
    {
      const __bf16* p = Plds + l16 * 32 + 8 * lh;
      pf.q[0] = *(const uint4*)(p);
      pf.q[1] = *(const uint4*)(p + 16);
    }
#pragma unroll
    for (int nb = 0; nb < 8; ++nb) {
      FragBF vf;
      const __bf16* p = Vt + (nb * 16 + l16) * 32 + 8 * lh;
      vf.q[0] = *(const uint4*)(p);
      vf.q[1] = *(const uint4*)(p + 16);
      o[nb] = __builtin_amdgcn_wmma_f32_16x16x32_bf16(false, pf.v, false, vf.v,
                                                      (short)0, o[nb], false, false);
    }
    __syncthreads();
  }

  // ---- normalize and store bf16 to aout[b][s][nh][hd]
#pragma unroll
  for (int r = 0; r < 8; ++r) rsum[r] = 1.f / rsum[r];
#pragma unroll
  for (int nb = 0; nb < 8; ++nb)
#pragma unroll
    for (int r = 0; r < 8; ++r) {
      const int m = r + 8 * lh;
      const int s = q0 + m;
      const size_t off = (((size_t)b * S_ + s) * NH_ + nh) * HD_ + nb * 16 + l16;
      aout[off] = (__bf16)(o[nb][r] * rsum[r]);
    }
}

// ---------------------------------------------------------------------------
// host launcher
// ---------------------------------------------------------------------------
extern "C" void kernel_launch(void* const* d_in, const int* in_sizes, int n_in,
                              void* d_out, int out_size, void* d_ws, size_t ws_size,
                              hipStream_t stream) {
  const float* x      = (const float*)d_in[0];
  const float* freqs  = (const float*)d_in[1];
  const float* w_qkv  = (const float*)d_in[2];
  const float* w_o    = (const float*)d_in[3];
  float* out = (float*)d_out;

  const size_t n_x    = (size_t)B_ * S_ * H_;        // 16,777,216
  const size_t n_wqkv = (size_t)3 * H_ * H_;         // 12,582,912
  const size_t n_wo   = (size_t)H_ * H_;             //  4,194,304

  __bf16* qkv    = (__bf16*)d_ws;                          // [3][B][NH][S][HD]
  __bf16* aout   = qkv + (size_t)3 * B_ * NH_ * S_ * HD_;  // [B][S][NH][HD]
  __bf16* xbf    = aout + (size_t)B_ * S_ * H_;
  __bf16* wqkvbf = xbf + n_x;
  __bf16* wobf   = wqkvbf + n_wqkv;

  const int M = B_ * S_;  // 8192

  // 0) pre-convert fp32 operands to bf16 (removes cvt from GEMM hot loops)
  f32_to_bf16_kernel<<<(unsigned)(n_x / 4 / 256), 256, 0, stream>>>(x, xbf, n_x / 4);
  f32_to_bf16_kernel<<<(unsigned)(n_wqkv / 4 / 256), 256, 0, stream>>>(w_qkv, wqkvbf, n_wqkv / 4);
  f32_to_bf16_kernel<<<(unsigned)(n_wo / 4 / 256), 256, 0, stream>>>(w_o, wobf, n_wo / 4);

  // 1) QKV projection: C[M, 3H] = x * w_qkv^T -> bf16 scatter into head layout
  dim3 g1(M / 64, (3 * H_) / 256);
  gemm_wmma_kernel<1><<<g1, 256, 0, stream>>>(xbf, wqkvbf, qkv, M, 3 * H_, H_);

  // 2) RoPE on q and k
  const size_t npairs = (size_t)2 * B_ * NH_ * S_ * (HD_ / 2);
  rope_kernel<<<(unsigned)((npairs + 255) / 256), 256, 0, stream>>>(qkv, freqs);

  // 3) causal flash attention
  dim3 g3(S_ / 16, NH_, B_);
  attn_kernel<<<g3, 32, 0, stream>>>(qkv, aout);

  // 4) output projection: out[M, H] = aout * w_o^T (fp32 store)
  dim3 g4(M / 64, H_ / 256);
  gemm_wmma_kernel<0><<<g4, 256, 0, stream>>>(aout, wobf, out, M, H_, H_);
}
